// Block_34067680592489
// MI455X (gfx1250) — matrix operand order repaired and checked by reading, not compiled, not run on hardware
//
#include <hip/hip_runtime.h>
#include <math.h>

// ---------------------------------------------------------------------------
// Types matching the gfx1250 WMMA builtin signatures
// ---------------------------------------------------------------------------
typedef __attribute__((ext_vector_type(16))) __bf16 v16bf;
typedef __attribute__((ext_vector_type(8)))  float  v8f;

union Frag {
    v16bf  v;
    __bf16 e[16];
    uint4  q[2];
};

__device__ __forceinline__ float softplus_f(float v) {
    // stable: max(v,0) + log1p(exp(-|v|))
    return fmaxf(v, 0.f) + log1pf(expf(-fabsf(v)));
}

// fp32 -> (bf16 hi via truncation, bf16 lo = round(v - hi)).  hi is exact
// (truncation), residual exactly representable: hi+lo carries ~16 mantissa
// bits, so the bf16x3 GEMM emulation reaches ~1e-5 relative accuracy.
__device__ __forceinline__ void splitf(float v, __bf16& h, __bf16& l) {
    unsigned u = __float_as_uint(v) & 0xffff0000u;
    h = __builtin_bit_cast(__bf16, (unsigned short)(u >> 16));
    l = (__bf16)(v - __uint_as_float(u));
}

// s_wait_asynccnt 0 : wait for this wave's async LDS loads to land
__device__ __forceinline__ void wait_async0() {
#if __has_builtin(__builtin_amdgcn_s_wait_asynccnt)
    __builtin_amdgcn_s_wait_asynccnt(0);
#else
    asm volatile("s_wait_asynccnt 0x0" ::: "memory");
#endif
}

// CDNA5 async global->LDS copy, 16B per lane, tracked with ASYNCcnt.
// lds_off: LDS byte address (low 32 bits of a generic pointer to __shared__).
__device__ __forceinline__ void async_copy_b128(unsigned lds_off, const void* gaddr) {
    asm volatile("global_load_async_to_lds_b128 %0, %1, off"
                 :: "v"(lds_off), "v"(gaddr)
                 : "memory");
}

// ---------------------------------------------------------------------------
// Kernel 1: split TtT / DtD (256x256 fp32) into bf16 hi/lo planes.
// Row-major along k == the exact B-fragment memory order for v_wmma bf16.
// ---------------------------------------------------------------------------
__global__ __launch_bounds__(256)
void prep_mats_kernel(const float* __restrict__ DtD, const float* __restrict__ TtT,
                      __bf16* __restrict__ Thi, __bf16* __restrict__ Tlo,
                      __bf16* __restrict__ Dhi, __bf16* __restrict__ Dlo) {
    int i = blockIdx.x * 256 + threadIdx.x;
    if (i < 256 * 256) {
        splitf(TtT[i], Thi[i], Tlo[i]);
        splitf(DtD[i], Dhi[i], Dlo[i]);
    }
}

// ---------------------------------------------------------------------------
// Kernel 2: per-row CNN -> mu[b], reg[b].  One wave32 per batch row, 8 rows
// per 256-thread block.  Row staged in LDS; uniform barriers between stages.
// ---------------------------------------------------------------------------
__global__ __launch_bounds__(256)
void cnn_kernel(const float* __restrict__ x,
                const float* __restrict__ w2m, const float* __restrict__ b2m,
                const float* __restrict__ w3m, const float* __restrict__ b3m,
                const float* __restrict__ lwm, const float* __restrict__ lbm,
                const float* __restrict__ w2r, const float* __restrict__ b2r,
                const float* __restrict__ w3r, const float* __restrict__ b3r,
                const float* __restrict__ lwr, const float* __restrict__ lbr,
                float* __restrict__ muArr, float* __restrict__ regArr) {
    __shared__ float xr[8][256];
    __shared__ float h1[8][64];
    __shared__ float h2[8][16];

    const int lane = threadIdx.x & 31;
    const int wv   = threadIdx.x >> 5;
    const int row  = blockIdx.x * 8 + wv;

    const float* xp = x + (size_t)row * 256 + lane * 8;
    *(float4*)&xr[wv][lane * 8]     = *(const float4*)(xp);
    *(float4*)&xr[wv][lane * 8 + 4] = *(const float4*)(xp + 4);
    __syncthreads();

    for (int s = 0; s < 2; ++s) {
        const float* w2 = s ? w2r : w2m;  const float b2 = (s ? b2r : b2m)[0];
        const float* w3 = s ? w3r : w3m;  const float b3 = (s ? b3r : b3m)[0];
        const float* lw = s ? lwr : lwm;  const float lb = (s ? lbr : lbm)[0];

        float wk[5];
        #pragma unroll
        for (int i = 0; i < 5; ++i) wk[i] = w2[i];

        // stage 1: conv5(pad2) over 256 -> avgpool4 -> softplus -> 64 vals
        #pragma unroll
        for (int half = 0; half < 2; ++half) {
            const int p = lane + half * 32;
            float acc = 0.f;
            #pragma unroll
            for (int t = 0; t < 4; ++t) {
                const int tt = 4 * p + t;
                #pragma unroll
                for (int i = 0; i < 5; ++i) {
                    const int k = tt - 2 + i;
                    const float xv = (k >= 0 && k < 256) ? xr[wv][k] : 0.f;
                    acc += wk[i] * xv;
                }
            }
            h1[wv][p] = softplus_f(b2 + 0.25f * acc);
        }
        __syncthreads();

        // stage 2: conv5(pad2) over 64 -> avgpool4 -> softplus -> 16 vals
        if (lane < 16) {
            float wk3[5];
            #pragma unroll
            for (int i = 0; i < 5; ++i) wk3[i] = w3[i];
            float acc = 0.f;
            #pragma unroll
            for (int t = 0; t < 4; ++t) {
                const int tt = 4 * lane + t;
                #pragma unroll
                for (int i = 0; i < 5; ++i) {
                    const int k = tt - 2 + i;
                    const float hv = (k >= 0 && k < 64) ? h1[wv][k] : 0.f;
                    acc += wk3[i] * hv;
                }
            }
            h2[wv][lane] = softplus_f(b3 + 0.25f * acc);
        }
        __syncthreads();

        // stage 3: linear 16->1 + softplus
        if (lane == 0) {
            float acc = lb;
            #pragma unroll
            for (int i = 0; i < 16; ++i) acc += lw[i] * h2[wv][i];
            (s ? regArr : muArr)[row] = softplus_f(acc);
        }
        __syncthreads();
    }
}

// ---------------------------------------------------------------------------
// Kernel 3: fused bf16x3 WMMA GEMM (TtT.x and DtD.x) + grad + Cardan solve.
// 8 waves/block, wave w owns rows [blk*128 + 16w, +16).  A-frags in regs,
// B tiles double-buffered in LDS via async global->LDS copies (ASYNCcnt),
// and B fragments ping-pong double-buffered in VGPRs so ds_load latency
// overlaps WMMA issue instead of forcing s_wait_dscnt 0 stalls.
// ---------------------------------------------------------------------------
#define NPAD 264   // row stride 528B: 16B-aligned, kills ds bank conflicts
#define BTILE (4 * 16 * NPAD)

__global__ __launch_bounds__(256)
void gemm_cardan_kernel(const float* __restrict__ x,
                        const float* __restrict__ x_b,
                        const __bf16* __restrict__ Thi,   // Tlo/Dhi/Dlo contiguous after
                        const float* __restrict__ muArr, const float* __restrict__ regArr,
                        const float* __restrict__ gamma_p, const float* __restrict__ massP,
                        float* __restrict__ out) {
    __shared__ __bf16 Bs[2][BTILE];

    const int tid   = threadIdx.x;
    const int lane  = tid & 31;
    const int wv    = tid >> 5;
    const int b0w   = blockIdx.x * 128 + wv * 16;
    const int arow  = b0w + (lane & 15);
    const int koffA = (lane & 16) ? 8 : 0;   // A-frag K split across lane halves

    // staging-copy coordinates: 256 threads move 4 mats x 16 rows x 256 k (32KB)
    const int sm  = tid >> 6;          // matrix plane: 0=Thi 1=Tlo 2=Dhi 3=Dlo
    const int t6  = tid & 63;
    const int snr = t6 >> 2;           // row in 16-row tile
    const int skq = (t6 & 3) * 64;     // k offset (bf16 elems)
    const __bf16* srcBase = Thi + (size_t)sm * 65536 + (size_t)snr * 256 + skq;
    const unsigned sOff   = (unsigned)(((sm * 16 + snr) * NPAD + skq) * 2);
    const unsigned bs0    = (unsigned)(uintptr_t)(void*)&Bs[0][0];
    const unsigned bs1    = (unsigned)(uintptr_t)(void*)&Bs[1][0];

    // async-stage tile jn (n0 = 16*jn) into buffer buf
    auto stage = [&](int jn, int buf) {
        const __bf16* src = srcBase + jn * (16 * 256);
        const unsigned dst = (buf ? bs1 : bs0) + sOff;
        #pragma unroll
        for (int q = 0; q < 8; ++q)
            async_copy_b128(dst + q * 16, (const void*)(src + q * 8));
    };

    stage(0, 0);   // prefetch first tile before doing anything else

    // ---- Load x rows once, split to bf16 hi/lo A fragments (kept in VGPRs).
    // 16-bit A 16x32 layout: lane<16: e[0..7]=K(c*32+0..7), e[8..15]=K(c*32+16..23);
    // lanes>=16 shifted by +8.
    Frag Ahi[8], Alo[8];
    const float* xrow = x + (size_t)arow * 256;
    #pragma unroll
    for (int c = 0; c < 8; ++c) {
        const int kb = c * 32 + koffA;
        float4 f0 = *(const float4*)(xrow + kb);
        float4 f1 = *(const float4*)(xrow + kb + 4);
        float4 f2 = *(const float4*)(xrow + kb + 16);
        float4 f3 = *(const float4*)(xrow + kb + 20);
        float fl[16] = {f0.x, f0.y, f0.z, f0.w, f1.x, f1.y, f1.z, f1.w,
                        f2.x, f2.y, f2.z, f2.w, f3.x, f3.y, f3.z, f3.w};
        #pragma unroll
        for (int i = 0; i < 16; ++i) splitf(fl[i], Ahi[c].e[i], Alo[c].e[i]);
    }

    const float gamma = softplus_f(gamma_p[0]);
    const float mass  = massP[0];

    const int nlane = lane & 15;
    const int koffB = (lane & 16) ? 8 : 0;   // B-frag: lanes>=16 shifted by +8 (K+8..15, K+24..31)

    // per-row scalars for the epilogue, hoisted out of the n-tile loop
    float muv[8], regv[8];
    #pragma unroll
    for (int r = 0; r < 8; ++r) {
        const int b = b0w + r + ((lane & 16) ? 8 : 0);
        muv[r]  = gamma * muArr[b];
        regv[r] = regArr[b];
    }

    for (int j = 0; j < 16; ++j) {
        const int jb = j & 1;
        const int n0 = j * 16;

        wait_async0();      // this wave's staged tile j is in LDS
        __syncthreads();    // whole block: tile j ready, tile j-1 reads done

        if (j + 1 < 16) stage(j + 1, jb ^ 1);   // overlap next tile with compute

        const __bf16* bb = &Bs[jb][0];

        // load the 4 B fragments (Thi/Tlo/Dhi/Dlo) of k-chunk c into fr[0..3]
        auto loadB = [&](Frag* fr, int c) {
            const int kb = c * 32 + koffB;
            #pragma unroll
            for (int m = 0; m < 4; ++m) {
                const __bf16* p = bb + (m * 16 + nlane) * NPAD + kb;
                fr[m].q[0] = *(const uint4*)(p);
                fr[m].q[1] = *(const uint4*)(p + 16);
            }
        };

        v8f accT = {};
        v8f accD = {};
        Frag fr[2][4];                 // ping-pong register double buffer
        loadB(&fr[0][0], 0);
        #pragma unroll
        for (int c = 0; c < 8; ++c) {
            if (c + 1 < 8) loadB(&fr[(c + 1) & 1][0], c + 1);  // prefetch next chunk
            Frag* f = &fr[c & 1][0];   // [0]=Thi [1]=Tlo [2]=Dhi [3]=Dlo
            // bf16x3: hi*hi + lo*hi + hi*lo, two independent accumulator chains
            accT = __builtin_amdgcn_wmma_f32_16x16x32_bf16(false, Ahi[c].v, false, f[0].v, (short)0, accT, false, false);
            accD = __builtin_amdgcn_wmma_f32_16x16x32_bf16(false, Ahi[c].v, false, f[2].v, (short)0, accD, false, false);
            accT = __builtin_amdgcn_wmma_f32_16x16x32_bf16(false, Alo[c].v, false, f[0].v, (short)0, accT, false, false);
            accD = __builtin_amdgcn_wmma_f32_16x16x32_bf16(false, Alo[c].v, false, f[2].v, (short)0, accD, false, false);
            accT = __builtin_amdgcn_wmma_f32_16x16x32_bf16(false, Ahi[c].v, false, f[1].v, (short)0, accT, false, false);
            accD = __builtin_amdgcn_wmma_f32_16x16x32_bf16(false, Ahi[c].v, false, f[3].v, (short)0, accD, false, false);
        }

        // ---- fused epilogue: grad -> x_tilde -> Cardan middle root.
        // C layout: VGPR r holds M = r (+8 for lanes>=16), N = lane&15.
        #pragma unroll
        for (int r = 0; r < 8; ++r) {
            const int b = b0w + r + ((lane & 16) ? 8 : 0);
            const int n = n0 + nlane;
            const size_t idx = (size_t)b * 256 + n;
            const float xv  = x[idx];
            const float xbv = x_b[idx];

            const float grad = accT[r] - xbv + regv[r] * accD[r];
            const float xt   = xv - gamma * grad;
            const float m    = muv[r];

            // cubic z^3 + cb z^2 + cc z + cd, middle trig root (always k=1)
            const float cb = -(mass + xt);
            const float cc = mass * xt - 2.f * m;
            const float cd = m * mass;
            float pp = cc - cb * cb * (1.f / 3.f);
            pp = fminf(pp, -1e-12f);
            const float qq = 2.f * cb * cb * cb * (1.f / 27.f) - cb * cc * (1.f / 3.f) + cd;
            const float rr = 2.f * sqrtf(-pp * (1.f / 3.f));
            float cosphi = 3.f * qq / (pp * rr);
            cosphi = fminf(fmaxf(cosphi, -1.f), 1.f);
            const float phi = acosf(cosphi);
            const float root =
                rr * cosf((phi - 6.28318530717958647692f) * (1.f / 3.f)) - cb * (1.f / 3.f);
            out[idx] = root;
        }
    }
}

// ---------------------------------------------------------------------------
// Launch
// ---------------------------------------------------------------------------
extern "C" void kernel_launch(void* const* d_in, const int* in_sizes, int n_in,
                              void* d_out, int out_size, void* d_ws, size_t ws_size,
                              hipStream_t stream) {
    const float* x       = (const float*)d_in[0];
    const float* x_b     = (const float*)d_in[1];
    const float* w2m     = (const float*)d_in[2];
    const float* b2m     = (const float*)d_in[3];
    const float* w3m     = (const float*)d_in[4];
    const float* b3m     = (const float*)d_in[5];
    const float* lwm     = (const float*)d_in[6];
    const float* lbm     = (const float*)d_in[7];
    const float* w2r     = (const float*)d_in[8];
    const float* b2r     = (const float*)d_in[9];
    const float* w3r     = (const float*)d_in[10];
    const float* b3r     = (const float*)d_in[11];
    const float* lwr     = (const float*)d_in[12];
    const float* lbr     = (const float*)d_in[13];
    const float* gamma_p = (const float*)d_in[14];
    const float* DtD     = (const float*)d_in[15];
    const float* TtT     = (const float*)d_in[16];
    // d_in[17] = U : unused by the reference
    const float* massP   = (const float*)d_in[18];
    float* out = (float*)d_out;

    const int B = in_sizes[0] / 256;   // 32768

    // workspace layout (~768 KB total): Thi|Tlo|Dhi|Dlo contiguous bf16 planes
    // (the GEMM kernel indexes them as Thi + plane*65536), then mu/reg.
    char* ws = (char*)d_ws;
    __bf16* Thi = (__bf16*)(ws + 0 * 131072);
    __bf16* Tlo = (__bf16*)(ws + 1 * 131072);
    __bf16* Dhi = (__bf16*)(ws + 2 * 131072);
    __bf16* Dlo = (__bf16*)(ws + 3 * 131072);
    float*  muA = (float*)(ws + 4 * 131072);
    float*  rgA = (float*)(ws + 4 * 131072 + (size_t)B * sizeof(float));

    prep_mats_kernel<<<256, 256, 0, stream>>>(DtD, TtT, Thi, Tlo, Dhi, Dlo);
    cnn_kernel<<<B / 8, 256, 0, stream>>>(x, w2m, b2m, w3m, b3m, lwm, lbm,
                                          w2r, b2r, w3r, b3r, lwr, lbr, muA, rgA);
    gemm_cardan_kernel<<<B / 128, 256, 0, stream>>>(x, x_b, Thi,
                                                    muA, rgA, gamma_p, massP, out);
}
